// IsoMax_91087666413905
// MI455X (gfx1250) — compile-verified, matrix operand order checked by reading, not run on hardware
//
#include <hip/hip_runtime.h>
#include <hip/hip_bf16.h>
#include <stdint.h>

// ---------------------------------------------------------------------------
// IsoMax logits: out[b][c] = -sqrt(max(||x_b||^2 + ||p_c||^2 - 2 x_b.p_c, 0))
// with x = features + 1e-6.
// Pass 1: split x and p into bf16 hi/lo planes (once) + fused row ||.||^2.
// Pass 2: WMMA bf16 GEMM (hi*hi + hi*lo + lo*hi) with async global->LDS fill.
// ---------------------------------------------------------------------------

#define BATCH 8192
#define NFEAT 256
#define NCLS  16384
#define EPS_F 1e-6f

#define BM 128
#define BN 128
#define BK 32
#define KCHUNKS (NFEAT / BK)   // 8

typedef __attribute__((ext_vector_type(16))) __bf16    v16bf;
typedef __attribute__((ext_vector_type(8)))  float     v8f;
typedef __attribute__((ext_vector_type(4)))  float     f32x4;
typedef __attribute__((ext_vector_type(4)))  uint32_t  u32x4;
typedef __attribute__((ext_vector_type(4)))  __bf16    bf16x4;
typedef int v4i __attribute__((vector_size(16)));      // matches builtin proto

union FragU { u32x4 q[2]; v16bf v; };

#if __has_builtin(__builtin_amdgcn_global_load_async_to_lds_b128)
#define HAS_ASYNC_LDS 1
#else
#define HAS_ASYNC_LDS 0
#endif

// 16B global -> LDS, async (ASYNCcnt) when available, else via VGPRs.
__device__ __forceinline__ void copy16_g2l(__bf16* ldst, const __bf16* gsrc) {
#if HAS_ASYNC_LDS
    __builtin_amdgcn_global_load_async_to_lds_b128(
        (__attribute__((address_space(1))) v4i*)gsrc,
        (__attribute__((address_space(3))) v4i*)ldst,
        /*imm offset=*/0, /*cpol=*/0);
#else
    *(u32x4*)ldst = *(const u32x4*)gsrc;
#endif
}

__device__ __forceinline__ void wait_async_done() {
#if HAS_ASYNC_LDS
#if __has_builtin(__builtin_amdgcn_s_wait_asynccnt)
    __builtin_amdgcn_s_wait_asynccnt(0);
#else
    asm volatile("s_wait_asynccnt 0" ::: "memory");
#endif
#endif
}

// Load one 16x32 bf16 fragment (A layout; B symmetric with lane = N):
//   lanes 0-15 : row = lane,    regs 0-3 -> K 0..7,  regs 4-7 -> K 16..23
//   lanes16-31 : row = lane-16, regs 0-3 -> K 8..15, regs 4-7 -> K 24..31
__device__ __forceinline__ v16bf load_frag(const __bf16* __restrict__ tile, int lane) {
    const int row = lane & 15;
    const int kb  = (lane & 16) ? 8 : 0;
    const __bf16* p = tile + row * BK;
    FragU f;
    f.q[0] = *(const u32x4*)(p + kb);        // ds_load_b128
    f.q[1] = *(const u32x4*)(p + kb + 16);
    return f.v;
}

__device__ __forceinline__ v8f wmma_bf16(v16bf a, v16bf b, v8f c) {
    return __builtin_amdgcn_wmma_f32_16x16x32_bf16(
        false, a, false, b, (short)0, c, false, false);
}

// ---------------------------------------------------------------------------
// Pass 1: per-row bf16 hi/lo split + sum of squares.  One wave per row.
// ---------------------------------------------------------------------------
__global__ __launch_bounds__(256) void split_rows_kernel(const float* __restrict__ X,
                                                         __bf16* __restrict__ Hi,
                                                         __bf16* __restrict__ Lo,
                                                         float* __restrict__ sq,
                                                         int nrows, float eps) {
    const int lane = threadIdx.x & 31;
    const int wid  = threadIdx.x >> 5;
    const int row  = blockIdx.x * 8 + wid;
    if (row >= nrows) return;
    const float* r  = X  + (size_t)row * NFEAT;
    __bf16*      hi = Hi + (size_t)row * NFEAT;
    __bf16*      lo = Lo + (size_t)row * NFEAT;
    float s = 0.0f;
#pragma unroll
    for (int half = 0; half < 2; ++half) {
        const int c = half * 128 + lane * 4;
        f32x4 v = *(const f32x4*)(r + c);
        bf16x4 h4, l4;
#pragma unroll
        for (int j = 0; j < 4; ++j) {
            float x = v[j] + eps;
            __bf16 h = (__bf16)x;             // RNE f32 -> bf16
            h4[j] = h;
            l4[j] = (__bf16)(x - (float)h);   // residual
            s = fmaf(x, x, s);
        }
        *(bf16x4*)(hi + c) = h4;
        *(bf16x4*)(lo + c) = l4;
    }
#pragma unroll
    for (int off = 16; off > 0; off >>= 1)
        s += __shfl_xor(s, off, 32);
    if (lane == 0) sq[row] = s;
}

// ---------------------------------------------------------------------------
// Pass 2: GEMM + distance epilogue.
// Block: 256 threads (8 waves), 128x128 tile, double-buffered LDS (64 KB).
// Wave tile: 64(M) x 32(N) = 4x2 WMMA tiles; 24 WMMA per K-chunk per wave.
// ---------------------------------------------------------------------------
__global__ __launch_bounds__(256) void isomax_kernel(const __bf16* __restrict__ Fhi,
                                                     const __bf16* __restrict__ Flo,
                                                     const __bf16* __restrict__ Phi,
                                                     const __bf16* __restrict__ Plo,
                                                     const float* __restrict__ xsq,
                                                     const float* __restrict__ psq,
                                                     float* __restrict__ out) {
    __shared__ __bf16 Ah[2][BM * BK];
    __shared__ __bf16 Al[2][BM * BK];
    __shared__ __bf16 Bh[2][BN * BK];
    __shared__ __bf16 Bl[2][BN * BK];

    const int tid  = threadIdx.x;
    const int lane = tid & 31;
    const int wid  = tid >> 5;

    const int bm = blockIdx.x * BM;       // feature-row base
    const int bn = blockIdx.y * BN;       // class base

    const int wm = (wid & 1) * 64;        // wave M offset
    const int wn = (wid >> 1) * 32;       // wave N offset

    v8f acc[4][2];
#pragma unroll
    for (int mt = 0; mt < 4; ++mt)
#pragma unroll
        for (int nt = 0; nt < 2; ++nt)
            acc[mt][nt] = (v8f)(0.0f);

    // Fill one K-chunk (4 planes x 8 KB).  Each thread moves 2 x 16B per
    // plane: seg = tid + 256*i -> row = seg>>2, col16 = seg&3.
    auto issue_chunk = [&](int kc, int buf) {
        const int k0 = kc * BK;
#pragma unroll
        for (int i = 0; i < 2; ++i) {
            const int seg = tid + 256 * i;
            const int r = seg >> 2;
            const int c = (seg & 3) * 8;                 // bf16 elements
            const size_t ga = (size_t)(bm + r) * NFEAT + k0 + c;
            const size_t gb = (size_t)(bn + r) * NFEAT + k0 + c;
            const int lo = r * BK + c;
            copy16_g2l(&Ah[buf][lo], Fhi + ga);
            copy16_g2l(&Al[buf][lo], Flo + ga);
            copy16_g2l(&Bh[buf][lo], Phi + gb);
            copy16_g2l(&Bl[buf][lo], Plo + gb);
        }
    };

    auto compute_chunk = [&](int buf) {
        v16bf bhf[2], blf[2];
#pragma unroll
        for (int nt = 0; nt < 2; ++nt) {
            bhf[nt] = load_frag(&Bh[buf][(wn + nt * 16) * BK], lane);
            blf[nt] = load_frag(&Bl[buf][(wn + nt * 16) * BK], lane);
        }
#pragma unroll
        for (int mt = 0; mt < 4; ++mt) {
            v16bf ah = load_frag(&Ah[buf][(wm + mt * 16) * BK], lane);
            v16bf al = load_frag(&Al[buf][(wm + mt * 16) * BK], lane);
#pragma unroll
            for (int nt = 0; nt < 2; ++nt) {
                acc[mt][nt] = wmma_bf16(ah, bhf[nt], acc[mt][nt]);  // hi*hi
                acc[mt][nt] = wmma_bf16(ah, blf[nt], acc[mt][nt]);  // hi*lo
                acc[mt][nt] = wmma_bf16(al, bhf[nt], acc[mt][nt]);  // lo*hi
            }
        }
    };

    issue_chunk(0, 0);
    wait_async_done();
    __syncthreads();
#pragma unroll 1
    for (int kc = 0; kc < KCHUNKS; ++kc) {
        const int cur = kc & 1;
        const bool more = (kc + 1) < KCHUNKS;
        if (more) issue_chunk(kc + 1, cur ^ 1);   // overlaps with compute
        compute_chunk(cur);
        wait_async_done();
        __syncthreads();
    }

    // Epilogue: C/D layout -> row = (lane>>4)*8 + reg, col = lane&15.
    const int nlane = lane & 15;
    const int mgrp  = (lane >> 4) * 8;
#pragma unroll
    for (int mt = 0; mt < 4; ++mt) {
#pragma unroll
        for (int nt = 0; nt < 2; ++nt) {
            const int col = bn + wn + nt * 16 + nlane;
            const float ps = psq[col];
#pragma unroll
            for (int r = 0; r < 8; ++r) {
                const int row = bm + wm + mt * 16 + mgrp + r;
                float sq = xsq[row] + ps - 2.0f * acc[mt][nt][r];
                sq = fmaxf(sq, 0.0f);
                out[(size_t)row * NCLS + col] = -__builtin_sqrtf(sq);
            }
        }
    }
}

// ---------------------------------------------------------------------------
extern "C" void kernel_launch(void* const* d_in, const int* in_sizes, int n_in,
                              void* d_out, int out_size, void* d_ws, size_t ws_size,
                              hipStream_t stream) {
    (void)in_sizes; (void)n_in; (void)out_size; (void)ws_size;
    const float* F = (const float*)d_in[0];   // features  [8192, 256]
    const float* P = (const float*)d_in[1];   // prototypes[16384, 256]
    float* out = (float*)d_out;               // [8192, 16384]

    // Workspace layout (24.1 MB total; planes are L2-resident: 24MB << 192MB)
    float*  xsq = (float*)d_ws;                            // [8192]
    float*  psq = xsq + BATCH;                             // [16384]
    __bf16* Fhi = (__bf16*)(psq + NCLS);                   // [8192 x 256]
    __bf16* Flo = Fhi + (size_t)BATCH * NFEAT;             // [8192 x 256]
    __bf16* Phi = Flo + (size_t)BATCH * NFEAT;             // [16384 x 256]
    __bf16* Plo = Phi + (size_t)NCLS * NFEAT;              // [16384 x 256]

    split_rows_kernel<<<BATCH / 8, 256, 0, stream>>>(F, Fhi, Flo, xsq, BATCH, EPS_F);
    split_rows_kernel<<<NCLS / 8, 256, 0, stream>>>(P, Phi, Plo, psq, NCLS, 0.0f);

    dim3 grid(BATCH / BM, NCLS / BN);         // 64 x 128 blocks
    isomax_kernel<<<grid, 256, 0, stream>>>(Fhi, Flo, Phi, Plo, xsq, psq, out);
}
